// TransformerModel_35141422416155
// MI455X (gfx1250) — matrix operand order repaired and checked by reading, not compiled, not run on hardware
//
#include <hip/hip_runtime.h>
#include <math.h>

typedef __attribute__((ext_vector_type(16))) _Float16 v16h;
typedef __attribute__((ext_vector_type(4)))  _Float16 v4h;
typedef __attribute__((ext_vector_type(8)))  float    v8f;

#define BM 128
#define BN 128
#define BK 32
#define LDSS 40              // half-element stride for LDS tiles (80B rows, 16B-aligned frag reads)

// ---------------- WMMA fragment builders (layouts per CDNA5 ISA 7.12.2) ------------

__device__ __forceinline__ v16h frag_a(const _Float16* As, int mbase, int lane) {
  // A is 16x32 (MxK). lane: m = lane&15 ; K-base = (lane>>4)*8
  // elements 0..7  -> K = kb..kb+7 ; elements 8..15 -> K = kb+16..kb+23
  const int m  = mbase + (lane & 15);
  const int kb = (lane >> 4) << 3;
  const _Float16* p = As + m * LDSS + kb;
  v16h a;
#pragma unroll
  for (int i = 0; i < 8; ++i) { a[i] = p[i]; a[8 + i] = p[16 + i]; }
  return a;
}

__device__ __forceinline__ v16h frag_b(const _Float16* Bs, int nbase, int lane) {
  // B is 32x16 (KxN), Bs stored [n][k]. lane: n = lane&15 ; K-base = (lane>>4)*8, elements 8..15 -> K+16
  const int n  = nbase + (lane & 15);
  const int kh = (lane >> 4) << 3;
  const _Float16* p = Bs + n * LDSS + kh;
  v16h b;
#pragma unroll
  for (int i = 0; i < 8; ++i) { b[i] = p[i]; b[8 + i] = p[16 + i]; }
  return b;
}

// ---------------- generic GEMM: C[M,ldc] = act(A[M,K] @ B[K,N] + bias) -------------
// A row-major lda=K, B row-major ldb=N. M%128==0, K%32==0 guaranteed by caller.
// act: 0 = none, 1 = exact GELU. Software-pipelined: next K-tile is prefetched into
// registers while WMMAs consume the current LDS tile.

__global__ __launch_bounds__(256) void gemm_f32_wmma(
    const float* __restrict__ A, const float* __restrict__ B,
    const float* __restrict__ bias, float* __restrict__ C,
    int M, int N, int K, int ldc, int act)
{
  __shared__ _Float16 As[BM * LDSS];
  __shared__ _Float16 Bs[BN * LDSS];

  const int tid  = threadIdx.x;
  const int lane = tid & 31;
  const int wave = tid >> 5;
  const int wm   = wave >> 1;       // 0..3 -> row tile 32*wm
  const int wn   = wave & 1;        // 0..1 -> col tile 64*wn
  const int bn0  = blockIdx.x * BN;
  const int bm0  = blockIdx.y * BM;

  v8f acc[2][4];
#pragma unroll
  for (int mi = 0; mi < 2; ++mi)
#pragma unroll
    for (int ni = 0; ni < 4; ++ni)
#pragma unroll
      for (int e = 0; e < 8; ++e) acc[mi][ni][e] = 0.0f;

  float4 areg[4], breg[4];

  // ---- tile fetch: global f32 -> registers (4 x float4 for A, 4 x float4 for B)
  auto ldg = [&](int k0) {
#pragma unroll
    for (int i = 0; i < 4; ++i) {                 // A tile 128x32
      const int idx = tid + i * 256;              // 0..1023
      const int r   = idx >> 3;
      const int kq  = (idx & 7) << 2;
      areg[i] = *(const float4*)(A + (size_t)(bm0 + r) * K + k0 + kq);
    }
#pragma unroll
    for (int i = 0; i < 4; ++i) {                 // B tile 32x128
      const int e  = (tid + i * 256) << 2;        // element 0..4095, step 4
      const int kk = e >> 7;
      const int n  = e & 127;
      const float* bp = B + (size_t)(k0 + kk) * N + bn0 + n;
      if (bn0 + n + 3 < N) {
        breg[i] = *(const float4*)bp;
      } else {
        float4 f;
        f.x = (bn0 + n + 0 < N) ? bp[0] : 0.0f;
        f.y = (bn0 + n + 1 < N) ? bp[1] : 0.0f;
        f.z = (bn0 + n + 2 < N) ? bp[2] : 0.0f;
        f.w = (bn0 + n + 3 < N) ? bp[3] : 0.0f;
        breg[i] = f;
      }
    }
  };

  // ---- registers -> LDS (f16 convert; B stored transposed [n][k])
  auto sts = [&]() {
#pragma unroll
    for (int i = 0; i < 4; ++i) {
      const int idx = tid + i * 256;
      const int r   = idx >> 3;
      const int kq  = (idx & 7) << 2;
      v4h h;
      h[0] = (_Float16)areg[i].x; h[1] = (_Float16)areg[i].y;
      h[2] = (_Float16)areg[i].z; h[3] = (_Float16)areg[i].w;
      *(v4h*)&As[r * LDSS + kq] = h;
    }
#pragma unroll
    for (int i = 0; i < 4; ++i) {
      const int e  = (tid + i * 256) << 2;
      const int kk = e >> 7;
      const int n  = e & 127;
      Bs[(n + 0) * LDSS + kk] = (_Float16)breg[i].x;
      Bs[(n + 1) * LDSS + kk] = (_Float16)breg[i].y;
      Bs[(n + 2) * LDSS + kk] = (_Float16)breg[i].z;
      Bs[(n + 3) * LDSS + kk] = (_Float16)breg[i].w;
    }
  };

  // ---- fragment loads + 8 WMMAs on current LDS tile
  auto compute = [&]() {
    v16h af0 = frag_a(As, wm * 32,      lane);
    v16h af1 = frag_a(As, wm * 32 + 16, lane);
    v16h bf0 = frag_b(Bs, wn * 64,      lane);
    v16h bf1 = frag_b(Bs, wn * 64 + 16, lane);
    v16h bf2 = frag_b(Bs, wn * 64 + 32, lane);
    v16h bf3 = frag_b(Bs, wn * 64 + 48, lane);
    acc[0][0] = __builtin_amdgcn_wmma_f32_16x16x32_f16(false, af0, false, bf0, (short)0, acc[0][0], false, false);
    acc[0][1] = __builtin_amdgcn_wmma_f32_16x16x32_f16(false, af0, false, bf1, (short)0, acc[0][1], false, false);
    acc[0][2] = __builtin_amdgcn_wmma_f32_16x16x32_f16(false, af0, false, bf2, (short)0, acc[0][2], false, false);
    acc[0][3] = __builtin_amdgcn_wmma_f32_16x16x32_f16(false, af0, false, bf3, (short)0, acc[0][3], false, false);
    acc[1][0] = __builtin_amdgcn_wmma_f32_16x16x32_f16(false, af1, false, bf0, (short)0, acc[1][0], false, false);
    acc[1][1] = __builtin_amdgcn_wmma_f32_16x16x32_f16(false, af1, false, bf1, (short)0, acc[1][1], false, false);
    acc[1][2] = __builtin_amdgcn_wmma_f32_16x16x32_f16(false, af1, false, bf2, (short)0, acc[1][2], false, false);
    acc[1][3] = __builtin_amdgcn_wmma_f32_16x16x32_f16(false, af1, false, bf3, (short)0, acc[1][3], false, false);
  };

  const int T = K / BK;
  ldg(0);
  sts();
  __syncthreads();
  for (int t = 1; t < T; ++t) {
    ldg(t * BK);        // prefetch next tile (overlaps with WMMAs below)
    compute();          // consume tile t-1 from LDS
    __syncthreads();    // all waves done reading LDS
    sts();              // write tile t (loadcnt wait lands here)
    __syncthreads();    // tile t visible
  }
  compute();            // last tile

  // epilogue: D layout -> m = (lane>>4)*8 + v, n = lane&15
  const int nlane = lane & 15;
  const int mhalf = lane >> 4;
#pragma unroll
  for (int mi = 0; mi < 2; ++mi) {
#pragma unroll
    for (int ni = 0; ni < 4; ++ni) {
      const int ncol = bn0 + wn * 64 + ni * 16 + nlane;
      if (ncol >= N) continue;
      const float bv = bias ? bias[ncol] : 0.0f;
#pragma unroll
      for (int v = 0; v < 8; ++v) {
        const int mrow = bm0 + wm * 32 + mi * 16 + mhalf * 8 + v;
        float val = acc[mi][ni][v] + bv;
        if (act == 1) val = 0.5f * val * (1.0f + erff(val * 0.70710678118f));
        C[(size_t)mrow * ldc + ncol] = val;
      }
    }
  }
}

// ---------------- chunked causal linear attention (per (b,head) block) -------------
// q,k,v,o: [B*S, 512] f32 ; head h occupies cols h*64..h*64+63.

__global__ __launch_bounds__(256) void attn_kernel(
    const float* __restrict__ q, const float* __restrict__ k,
    const float* __restrict__ v, float* __restrict__ o)
{
  __shared__ float Sst[64][65];
  __shared__ float zv[64];
  __shared__ float Qc[16][64];
  __shared__ float Kc[16][64];
  __shared__ float Vc[16][64];
  __shared__ float Am[16][16];
  __shared__ float den[16];

  const int tid  = threadIdx.x;
  const int b    = blockIdx.x >> 3;
  const int head = blockIdx.x & 7;
  const size_t rowbase = (size_t)b * 1024;
  const int cbase = head * 64;

  for (int i = tid; i < 64 * 64; i += 256) Sst[i >> 6][i & 63] = 0.0f;
  if (tid < 64) zv[tid] = 0.0f;
  __syncthreads();

  for (int s0 = 0; s0 < 1024; s0 += 16) {
    // load chunk, apply elu(x)+1 feature map to q,k
#pragma unroll
    for (int i = 0; i < 4; ++i) {
      const int idx = tid + i * 256;
      const int rr = idx >> 6, d = idx & 63;
      const size_t g = (rowbase + s0 + rr) * 512 + cbase + d;
      const float qq = q[g], kk = k[g];
      Qc[rr][d] = qq > 0.0f ? qq + 1.0f : __expf(qq);
      Kc[rr][d] = kk > 0.0f ? kk + 1.0f : __expf(kk);
      Vc[rr][d] = v[g];
    }
    __syncthreads();

    // A = (Qc Kc^T) with causal mask (j<=i)
    {
      const int i = tid >> 4, j = tid & 15;
      float a = 0.0f;
      if (j <= i) {
#pragma unroll
        for (int d = 0; d < 64; ++d) a += Qc[i][d] * Kc[j][d];
      }
      Am[i][j] = a;
    }
    __syncthreads();

    // denominator: q . (z_prev + cumsum_k) = Qc.z_prev + rowsum(A)
    if (tid < 16) {
      float dsum = 1e-6f;
#pragma unroll
      for (int j = 0; j < 16; ++j) dsum += Am[tid][j];
#pragma unroll
      for (int d = 0; d < 64; ++d) dsum += Qc[tid][d] * zv[d];
      den[tid] = dsum;
    }
    __syncthreads();

    // numerator = Qc @ S_prev + A @ Vc ; write output
#pragma unroll
    for (int i4 = 0; i4 < 4; ++i4) {
      const int idx = tid + i4 * 256;
      const int rr = idx >> 6, dv = idx & 63;
      float num = 0.0f;
#pragma unroll 8
      for (int d = 0; d < 64; ++d) num += Qc[rr][d] * Sst[d][dv];
#pragma unroll
      for (int j = 0; j < 16; ++j) num += Am[rr][j] * Vc[j][dv];
      o[(rowbase + s0 + rr) * 512 + cbase + dv] = num / den[rr];
    }
    __syncthreads();

    // state update: S += Kc^T Vc ; z += sum_j Kc[j]
#pragma unroll
    for (int i16 = 0; i16 < 16; ++i16) {
      const int idx = tid + i16 * 256;
      const int dk = idx >> 6, dv = idx & 63;
      float a = 0.0f;
#pragma unroll
      for (int j = 0; j < 16; ++j) a += Kc[j][dk] * Vc[j][dv];
      Sst[dk][dv] += a;
    }
    if (tid < 64) {
      float a = 0.0f;
#pragma unroll
      for (int j = 0; j < 16; ++j) a += Kc[j][tid];
      zv[tid] += a;
    }
    __syncthreads();
  }
}

// ---------------- fused residual + LayerNorm (D=512, one row per block) ------------

__global__ __launch_bounds__(256) void ln_kernel(
    const float* __restrict__ x, const float* __restrict__ resid,
    const float* __restrict__ g, const float* __restrict__ b,
    float* __restrict__ out)
{
  __shared__ float red[256];
  const int row = blockIdx.x, tid = threadIdx.x;
  const size_t base = (size_t)row * 512;
  float v0 = x[base + tid], v1 = x[base + tid + 256];
  if (resid) { v0 += resid[base + tid]; v1 += resid[base + tid + 256]; }
  red[tid] = v0 + v1;
  __syncthreads();
  for (int ofs = 128; ofs > 0; ofs >>= 1) { if (tid < ofs) red[tid] += red[tid + ofs]; __syncthreads(); }
  const float mean = red[0] * (1.0f / 512.0f);
  __syncthreads();
  const float d0 = v0 - mean, d1 = v1 - mean;
  red[tid] = d0 * d0 + d1 * d1;
  __syncthreads();
  for (int ofs = 128; ofs > 0; ofs >>= 1) { if (tid < ofs) red[tid] += red[tid + ofs]; __syncthreads(); }
  const float rstd = rsqrtf(red[0] * (1.0f / 512.0f) + 1e-5f);
  out[base + tid]       = d0 * rstd * g[tid]       + b[tid];
  out[base + tid + 256] = d1 * rstd * g[tid + 256] + b[tid + 256];
}

// ---------------- embeddings gather + scale ---------------------------------------

__global__ __launch_bounds__(256) void embed_kernel(
    const int* __restrict__ x,
    const float* __restrict__ t0, const float* __restrict__ t1,
    const float* __restrict__ t2, const float* __restrict__ t3,
    const float* __restrict__ t4, const float* __restrict__ t5,
    const float* __restrict__ t6, float* __restrict__ embs)
{
  const int r = blockIdx.x;
  const int* xr = x + r * 7;
  float* er = embs + (size_t)r * 1248;
  for (int c = threadIdx.x; c < 1248; c += 256) {
    const float* t; int off, dim, idx; float sc;
    if      (c < 128)  { t = t0; off = 0;    dim = 128; sc = 11.3137085f;  idx = xr[0]; }
    else if (c < 384)  { t = t1; off = 128;  dim = 256; sc = 16.0f;        idx = xr[1]; }
    else if (c < 448)  { t = t2; off = 384;  dim = 64;  sc = 8.0f;         idx = xr[2]; }
    else if (c < 480)  { t = t3; off = 448;  dim = 32;  sc = 5.65685425f;  idx = xr[3]; }
    else if (c < 992)  { t = t4; off = 480;  dim = 512; sc = 22.627417f;   idx = xr[4]; }
    else if (c < 1120) { t = t5; off = 992;  dim = 128; sc = 11.3137085f;  idx = xr[5]; }
    else               { t = t6; off = 1120; dim = 128; sc = 11.3137085f;  idx = xr[6]; }
    er[c] = t[idx * dim + (c - off)] * sc;
  }
}

// ---------------- sinusoidal positional encoding add ------------------------------

__global__ __launch_bounds__(256) void posenc_kernel(float* __restrict__ h)
{
  const int r = blockIdx.x, j = threadIdx.x;
  const int s = r & 1023;
  const float div = __expf((float)(2 * j) * (-9.210340372f / 512.0f));
  const float a = (float)s * div;
  h[(size_t)r * 512 + 2 * j]     += __sinf(a);
  h[(size_t)r * 512 + 2 * j + 1] += __cosf(a);
}

// ---------------- concat [h, type_emb(y)] -----------------------------------------

__global__ __launch_bounds__(256) void concat_kernel(
    const float* __restrict__ h, const int* __restrict__ y,
    const float* __restrict__ emb_type, float* __restrict__ c)
{
  const int r = blockIdx.x;
  const int ti = y[r * 7 + 3];
  float* cr = c + (size_t)r * 544;
  const float* hr = h + (size_t)r * 512;
  for (int j = threadIdx.x; j < 544; j += 256)
    cr[j] = (j < 512) ? hr[j] : emb_type[ti * 32 + (j - 512)] * 5.65685425f;
}

// ---------------- host-side orchestration -----------------------------------------

static inline void gemm(hipStream_t st, const float* A, const float* B, const float* bias,
                        float* C, int M, int N, int K, int ldc, int act)
{
  dim3 grid((N + BN - 1) / BN, M / BM);
  gemm_f32_wmma<<<grid, 256, 0, st>>>(A, B, bias, C, M, N, K, ldc, act);
}

extern "C" void kernel_launch(void* const* d_in, const int* in_sizes, int n_in,
                              void* d_out, int out_size, void* d_ws, size_t ws_size,
                              hipStream_t stream)
{
  (void)in_sizes; (void)n_in; (void)out_size; (void)ws_size;
  const int*   x        = (const int*)  d_in[0];
  const int*   y        = (const int*)  d_in[1];
  const float* e_tempo  = (const float*)d_in[2];
  const float* e_chord  = (const float*)d_in[3];
  const float* e_barb   = (const float*)d_in[4];
  const float* e_type   = (const float*)d_in[5];
  const float* e_pitch  = (const float*)d_in[6];
  const float* e_dur    = (const float*)d_in[7];
  const float* e_vel    = (const float*)d_in[8];
  const float* in_w     = (const float*)d_in[9];
  const float* in_b     = (const float*)d_in[10];
  const float* wq       = (const float*)d_in[11];
  const float* wk       = (const float*)d_in[12];
  const float* wv       = (const float*)d_in[13];
  const float* wo       = (const float*)d_in[14];
  const float* bq       = (const float*)d_in[15];
  const float* bk       = (const float*)d_in[16];
  const float* bv       = (const float*)d_in[17];
  const float* bo       = (const float*)d_in[18];
  const float* ln1_g    = (const float*)d_in[19];
  const float* ln1_b    = (const float*)d_in[20];
  const float* ln2_g    = (const float*)d_in[21];
  const float* ln2_b    = (const float*)d_in[22];
  const float* ff_w1    = (const float*)d_in[23];
  const float* ff_b1    = (const float*)d_in[24];
  const float* ff_w2    = (const float*)d_in[25];
  const float* ff_b2    = (const float*)d_in[26];
  const float* lnf_g    = (const float*)d_in[27];
  const float* lnf_b    = (const float*)d_in[28];
  const float* pc_w     = (const float*)d_in[29];
  const float* pc_b     = (const float*)d_in[30];
  const float* pw[7] = { (const float*)d_in[31], (const float*)d_in[33], (const float*)d_in[35],
                         (const float*)d_in[37], (const float*)d_in[39], (const float*)d_in[41],
                         (const float*)d_in[43] };
  const float* pb[7] = { (const float*)d_in[32], (const float*)d_in[34], (const float*)d_in[36],
                         (const float*)d_in[38], (const float*)d_in[40], (const float*)d_in[42],
                         (const float*)d_in[44] };

  const int M = 4096;                 // B*S
  char* ws = (char*)d_ws;
  // region reuse: BIG holds embs, then ff intermediate, then concat buffer
  float* big  = (float*)(ws);                         // 32 MB (4096*2048*4)
  float* hbuf = (float*)(ws + 33554432);              // 8 MB each below
  float* qbuf = (float*)(ws + 33554432 + 1 * 8388608);
  float* kbuf = (float*)(ws + 33554432 + 2 * 8388608);
  float* vbuf = (float*)(ws + 33554432 + 3 * 8388608);
  float* abuf = (float*)(ws + 33554432 + 4 * 8388608);
  float* obuf = (float*)(ws + 33554432 + 5 * 8388608);
  float* out  = (float*)d_out;                        // [4096, 343]

  // 1) embeddings -> big[4096,1248] ; h = embs @ in_w + in_b ; h += posenc
  embed_kernel<<<M, 256, 0, stream>>>(x, e_tempo, e_chord, e_barb, e_type, e_pitch, e_dur, e_vel, big);
  gemm(stream, big, in_w, in_b, hbuf, M, 512, 1248, 512, 0);
  posenc_kernel<<<M, 256, 0, stream>>>(hbuf);

  // 2) 12 transformer layers
  for (int l = 0; l < 12; ++l) {
    const size_t wof = (size_t)l * 512 * 512;
    gemm(stream, hbuf, wq + wof, bq + l * 512, qbuf, M, 512, 512, 512, 0);
    gemm(stream, hbuf, wk + wof, bk + l * 512, kbuf, M, 512, 512, 512, 0);
    gemm(stream, hbuf, wv + wof, bv + l * 512, vbuf, M, 512, 512, 512, 0);
    attn_kernel<<<32, 256, 0, stream>>>(qbuf, kbuf, vbuf, abuf);
    gemm(stream, abuf, wo + wof, bo + l * 512, obuf, M, 512, 512, 512, 0);
    ln_kernel<<<M, 256, 0, stream>>>(hbuf, obuf, ln1_g + l * 512, ln1_b + l * 512, hbuf);
    gemm(stream, hbuf, ff_w1 + (size_t)l * 512 * 2048, ff_b1 + l * 2048, big, M, 2048, 512, 2048, 1);
    gemm(stream, big, ff_w2 + (size_t)l * 2048 * 512, ff_b2 + l * 512, obuf, M, 512, 2048, 512, 0);
    ln_kernel<<<M, 256, 0, stream>>>(hbuf, obuf, ln2_g + l * 512, ln2_b + l * 512, hbuf);
  }

  // 3) final LN
  ln_kernel<<<M, 256, 0, stream>>>(hbuf, nullptr, lnf_g, lnf_b, hbuf);

  // 4) y_type = h @ proj_type (-> out cols 209..212)
  gemm(stream, hbuf, pw[3], pb[3], out + 209, M, 4, 512, 343, 0);

  // 5) y_ = concat(h, type_emb(y)) @ pc_w + pc_b   (concat buffer reuses BIG region)
  concat_kernel<<<M, 256, 0, stream>>>(hbuf, y, e_type, big);
  gemm(stream, big, pc_w, pc_b, qbuf, M, 512, 544, 512, 0);   // ybuf reuses qbuf

  // 6) remaining projections into output slices (ldc = 343)
  const int pn[7]   = { 56, 135, 18, 4, 87, 18, 25 };
  const int pofs[7] = { 0, 56, 191, 209, 213, 300, 318 };
  for (int i = 0; i < 7; ++i) {
    if (i == 3) continue;  // type head already done from h
    gemm(stream, qbuf, pw[i], pb[i], out + pofs[i], M, pn[i], 512, 343, 0);
  }
}
